// PaperClassifier_33724083208437
// MI455X (gfx1250) — matrix operand order repaired and checked by reading, not compiled, not run on hardware
//
#include <hip/hip_runtime.h>

typedef __attribute__((ext_vector_type(16))) __bf16 v16bf;
typedef __attribute__((ext_vector_type(8)))  float  v8f;

// ---------------------------------------------------------------------------
// Degree kernels:  deg = segment_sum(ones, dst); dinv = rsqrt(deg+1),
// invdeg = 1/(deg+1)  (self-loop included as +1, matching the reference)
// ---------------------------------------------------------------------------
__global__ void k_deg_count(const int* __restrict__ dst, float* __restrict__ deg, int E) {
    int e = blockIdx.x * blockDim.x + threadIdx.x;
    if (e < E) atomicAdd(&deg[dst[e]], 1.0f);
}

__global__ void k_deg_finalize(const float* __restrict__ deg, float* __restrict__ dinv,
                               float* __restrict__ invdeg, int N) {
    int i = blockIdx.x * blockDim.x + threadIdx.x;
    if (i < N) {
        float d = deg[i] + 1.0f;
        dinv[i]   = rsqrtf(d);
        invdeg[i] = 1.0f / d;
    }
}

// ---------------------------------------------------------------------------
// bf16x3 split: x = hi + lo (each bf16).  Plain row-major, done ONCE per layer
// (the old kernel re-converted every A element 16x inside the GEMM).
// ---------------------------------------------------------------------------
__global__ void k_split_bf16(const float* __restrict__ x, __bf16* __restrict__ hi,
                             __bf16* __restrict__ lo, long long n) {
    long long i = (long long)blockIdx.x * blockDim.x + threadIdx.x;
    if (i < n) {
        float v  = x[i];
        __bf16 h = (__bf16)v;
        hi[i] = h;
        lo[i] = (__bf16)(v - (float)h);
    }
}

// ---------------------------------------------------------------------------
// Pack weights B[K,Nc] into exact WMMA B-fragment order (hi/lo), zero-padded
// to a multiple of 16 columns: frag[(tileN*Kt + kt)*32 + lane][j], where
// element j of lane {half,lidx} is B[kt*32 + half*8 + (j&7) + 16*(j>>3)][tileN*16 + lidx].
// One contiguous 32B v16bf load per wave per K-step in the GEMM.
// grid = (Kt, Ntiles), block = 32.
// ---------------------------------------------------------------------------
__global__ void k_pack_b(const float* __restrict__ B, __bf16* __restrict__ bhi,
                         __bf16* __restrict__ blo, int K, int Nc) {
    int kt = blockIdx.x, tn = blockIdx.y, lane = threadIdx.x;
    int half = lane >> 4, lidx = lane & 15;
    int col  = tn * 16 + lidx;
    int Kt   = K >> 5;
    size_t base = (((size_t)tn * Kt + kt) * 32 + lane) * 16;
#pragma unroll
    for (int j = 0; j < 16; ++j) {
        int   kk = kt * 32 + half * 8 + (j & 7) + ((j >> 3) << 4);
        float v  = (col < Nc) ? B[(size_t)kk * Nc + col] : 0.0f;
        __bf16 h = (__bf16)v;
        bhi[base + j] = h;
        blo[base + j] = (__bf16)(v - (float)h);
    }
}

// ---------------------------------------------------------------------------
// GEMM  C[M,Nc] = A[M,K] @ B[K,Nc]  via bf16x3 (hi*hi + hi*lo + lo*hi, fp32 acc).
// A: pre-split row-major bf16 hi/lo.  B: pre-packed fragments.
// Inner loop = 8x global_load_b128 + 3x v_wmma_f32_16x16x32_bf16, no guards,
// no 64-bit multiplies, no conversions.
// A fragment (ISA 7.12.2): lane={half,m}: j<8 -> K=k0+half*8+j ; j>=8 -> +16.
// C/D: element r -> row m0 + r + 8*half, col n0 + (lane&15).
// ---------------------------------------------------------------------------
__global__ void k_gemm_bf16x3(const __bf16* __restrict__ Ahi, const __bf16* __restrict__ Alo,
                              const __bf16* __restrict__ Bhi, const __bf16* __restrict__ Blo,
                              float* __restrict__ C, int M, int K, int Nc) {
    const int Mtiles = M >> 4;
    const int Kt     = K >> 5;
    const int tileM  = blockIdx.x * blockDim.y + threadIdx.y;
    if (tileM >= Mtiles) return;               // uniform per wave: EXEC all-1s for WMMA
    const int tileN = blockIdx.y;
    const int lane  = threadIdx.x;
    const int half  = lane >> 4;
    const int lidx  = lane & 15;
    const int m0 = tileM << 4, n0 = tileN << 4;
    const int col = n0 + lidx;

    const __bf16* __restrict__ a0 = Ahi + (size_t)(m0 + lidx) * K + half * 8;
    const __bf16* __restrict__ a1 = Alo + (size_t)(m0 + lidx) * K + half * 8;
    const __bf16* __restrict__ b0 = Bhi + ((size_t)tileN * Kt * 32 + lane) * 16;
    const __bf16* __restrict__ b1 = Blo + ((size_t)tileN * Kt * 32 + lane) * 16;

    v8f acc = {};
    for (int kt = 0; kt < Kt; ++kt) {
        union { uint4 q[2]; v16bf v; } ah, al;
        ah.q[0] = *(const uint4*)(a0);          // K = k0+half*8 .. +7      (16B)
        ah.q[1] = *(const uint4*)(a0 + 16);     // K = k0+16+half*8 .. +7   (16B)
        al.q[0] = *(const uint4*)(a1);
        al.q[1] = *(const uint4*)(a1 + 16);
        v16bf bh = *(const v16bf*)(b0);         // contiguous 32B fragment
        v16bf bl = *(const v16bf*)(b1);

        acc = __builtin_amdgcn_wmma_f32_16x16x32_bf16(false, ah.v, false, bl,
                                                      (short)0, acc, false, false);
        acc = __builtin_amdgcn_wmma_f32_16x16x32_bf16(false, al.v, false, bh,
                                                      (short)0, acc, false, false);
        acc = __builtin_amdgcn_wmma_f32_16x16x32_bf16(false, ah.v, false, bh,
                                                      (short)0, acc, false, false);
        a0 += 32; a1 += 32; b0 += 512; b1 += 512;
    }

    if (col < Nc) {
        float* __restrict__ crow = C + (size_t)(m0 + 8 * half) * Nc + col;
#pragma unroll
        for (int r = 0; r < 8; ++r) crow[(size_t)r * Nc] = acc[r];
    }
}

// ---------------------------------------------------------------------------
// Edge scatter: agg[dst] += dinv[src]*dinv[dst] * h[src]   (L2-resident atomics)
// ---------------------------------------------------------------------------
__global__ void k_scatter(const float* __restrict__ h, float* __restrict__ agg,
                          const int* __restrict__ src, const int* __restrict__ dst,
                          const float* __restrict__ dinv, int H) {
    int e = blockIdx.x;
    int s = src[e], d = dst[e];
    float norm = dinv[s] * dinv[d];
    const float* __restrict__ hs = h   + (size_t)s * H;
    float*       __restrict__ ad = agg + (size_t)d * H;
    for (int f = threadIdx.x; f < H; f += blockDim.x)
        atomicAdd(&ad[f], hs[f] * norm);
}

// out = agg + h * invdeg[row] + bias[col]
__global__ void k_combine(const float* __restrict__ agg, const float* __restrict__ h,
                          const float* __restrict__ invdeg, const float* __restrict__ bias,
                          float* __restrict__ out, long long total, int H) {
    long long idx = (long long)blockIdx.x * blockDim.x + threadIdx.x;
    if (idx >= total) return;
    int i = (int)(idx / H);
    int f = (int)(idx - (long long)i * H);
    out[idx] = agg[idx] + h[idx] * invdeg[i] + bias[f];
}

// ---------------------------------------------------------------------------
// BatchNorm (training mode, H==256): thread t owns feature t -> coalesced.
// ---------------------------------------------------------------------------
__global__ void k_bn_stats(const float* __restrict__ x, float* __restrict__ stats, int N) {
    int f    = threadIdx.x;
    int row0 = blockIdx.x * 64;
    int rend = row0 + 64; if (rend > N) rend = N;
    float s1 = 0.0f, s2 = 0.0f;
    for (int i = row0; i < rend; ++i) {
        float v = x[(size_t)i * 256 + f];
        s1 += v; s2 += v * v;
    }
    atomicAdd(&stats[f],       s1);
    atomicAdd(&stats[256 + f], s2);
}

__global__ void k_bn_finalize(const float* __restrict__ stats, const float* __restrict__ g,
                              const float* __restrict__ be, float* __restrict__ scale,
                              float* __restrict__ shift, float invN) {
    int f = threadIdx.x;
    float mean = stats[f] * invN;
    float var  = stats[256 + f] * invN - mean * mean;   // biased variance
    float sc   = g[f] * rsqrtf(var + 1e-5f);
    scale[f] = sc;
    shift[f] = be[f] - mean * sc;
}

__global__ void k_bn_apply_relu(const float* __restrict__ x, const float* __restrict__ scale,
                                const float* __restrict__ shift, float* __restrict__ y,
                                long long total) {
    long long idx = (long long)blockIdx.x * blockDim.x + threadIdx.x;
    if (idx >= total) return;
    int f = (int)(idx & 255);
    float v = x[idx] * scale[f] + shift[f];
    y[idx] = v > 0.0f ? v : 0.0f;
}

// ---------------------------------------------------------------------------
static inline size_t alignup(size_t v) { return (v + 255) & ~(size_t)255; }

extern "C" void kernel_launch(void* const* d_in, const int* in_sizes, int n_in,
                              void* d_out, int out_size, void* d_ws, size_t ws_size,
                              hipStream_t stream) {
    const float* x   = (const float*)d_in[0];
    const int*   ei  = (const int*)d_in[1];    // edge_index (integer -> const int*)
    const float* W1  = (const float*)d_in[2];
    const float* b1  = (const float*)d_in[3];
    const float* W2  = (const float*)d_in[4];
    const float* b2  = (const float*)d_in[5];
    const float* W3  = (const float*)d_in[6];
    const float* b3  = (const float*)d_in[7];
    const float* g1  = (const float*)d_in[8];
    const float* be1 = (const float*)d_in[9];
    const float* g2  = (const float*)d_in[10];
    const float* be2 = (const float*)d_in[11];

    const int H   = in_sizes[3];               // 256
    const int DIN = in_sizes[2] / H;           // 384
    const int N   = in_sizes[0] / DIN;         // 50000
    const int E   = in_sizes[1] / 2;           // 800000
    const int C   = in_sizes[7];               // 10
    const int* srcI = ei;
    const int* dstI = ei + E;

    // ---- workspace carve-out ----
    char* ws = (char*)d_ws;
    size_t off = 0;
    float*  deg    = (float*)(ws + off); off = alignup(off + (size_t)N * 4);
    float*  dinv   = (float*)(ws + off); off = alignup(off + (size_t)N * 4);
    float*  invdeg = (float*)(ws + off); off = alignup(off + (size_t)N * 4);
    float*  stats  = (float*)(ws + off); off = alignup(off + 512 * 4);
    float*  scale  = (float*)(ws + off); off = alignup(off + 256 * 4);
    float*  shift  = (float*)(ws + off); off = alignup(off + 256 * 4);
    __bf16* bfh    = (__bf16*)(ws + off); off = alignup(off + (size_t)DIN * H * 2); // packed W hi
    __bf16* bfl    = (__bf16*)(ws + off); off = alignup(off + (size_t)DIN * H * 2); // packed W lo
    __bf16* ahi    = (__bf16*)(ws + off); off = alignup(off + (size_t)N * DIN * 2); // split A hi
    __bf16* alo    = (__bf16*)(ws + off); off = alignup(off + (size_t)N * DIN * 2); // split A lo
    float*  big0   = (float*)(ws + off); off = alignup(off + (size_t)N * H * 4);    // layer input
    float*  big1   = (float*)(ws + off); off = alignup(off + (size_t)N * H * 4);    // h = x@W
    float*  big2   = (float*)(ws + off); off = alignup(off + (size_t)N * H * 4);    // agg / out

    const long long totH = (long long)N * H;
    const long long totC = (long long)N * C;
    const int Mtiles = N >> 4;                 // 3125
    const dim3 gblock(32, 4);
    const unsigned gx = (unsigned)((Mtiles + 3) / 4);
    const int NtH = H / 16, NtC = (C + 15) / 16;

    // ---- degrees ----
    hipMemsetAsync(deg, 0, (size_t)N * 4, stream);
    k_deg_count<<<(E + 255) / 256, 256, 0, stream>>>(dstI, deg, E);
    k_deg_finalize<<<(N + 255) / 256, 256, 0, stream>>>(deg, dinv, invdeg, N);

    // ================= Layer 1: GCNConv(384->256) + BN + ReLU =================
    {
        long long nA = (long long)N * DIN;
        k_split_bf16<<<(unsigned)((nA + 255) / 256), 256, 0, stream>>>(x, ahi, alo, nA);
        k_pack_b<<<dim3(DIN / 32, NtH), 32, 0, stream>>>(W1, bfh, bfl, DIN, H);
        k_gemm_bf16x3<<<dim3(gx, NtH), gblock, 0, stream>>>(ahi, alo, bfh, bfl, big1, N, DIN, H);
    }
    hipMemsetAsync(big2, 0, (size_t)totH * 4, stream);
    k_scatter<<<E, 256, 0, stream>>>(big1, big2, srcI, dstI, dinv, H);
    k_combine<<<(unsigned)((totH + 255) / 256), 256, 0, stream>>>(big2, big1, invdeg, b1,
                                                                  big2, totH, H);
    hipMemsetAsync(stats, 0, 512 * 4, stream);
    k_bn_stats<<<(N + 63) / 64, 256, 0, stream>>>(big2, stats, N);
    k_bn_finalize<<<1, 256, 0, stream>>>(stats, g1, be1, scale, shift, 1.0f / (float)N);
    k_bn_apply_relu<<<(unsigned)((totH + 255) / 256), 256, 0, stream>>>(big2, scale, shift,
                                                                        big0, totH);

    // ================= Layer 2: GCNConv(256->256) + BN + ReLU =================
    {
        long long nA = totH;
        k_split_bf16<<<(unsigned)((nA + 255) / 256), 256, 0, stream>>>(big0, ahi, alo, nA);
        k_pack_b<<<dim3(H / 32, NtH), 32, 0, stream>>>(W2, bfh, bfl, H, H);
        k_gemm_bf16x3<<<dim3(gx, NtH), gblock, 0, stream>>>(ahi, alo, bfh, bfl, big1, N, H, H);
    }
    hipMemsetAsync(big2, 0, (size_t)totH * 4, stream);
    k_scatter<<<E, 256, 0, stream>>>(big1, big2, srcI, dstI, dinv, H);
    k_combine<<<(unsigned)((totH + 255) / 256), 256, 0, stream>>>(big2, big1, invdeg, b2,
                                                                  big2, totH, H);
    hipMemsetAsync(stats, 0, 512 * 4, stream);
    k_bn_stats<<<(N + 63) / 64, 256, 0, stream>>>(big2, stats, N);
    k_bn_finalize<<<1, 256, 0, stream>>>(stats, g2, be2, scale, shift, 1.0f / (float)N);
    k_bn_apply_relu<<<(unsigned)((totH + 255) / 256), 256, 0, stream>>>(big2, scale, shift,
                                                                        big0, totH);

    // ================= Layer 3: GCNConv(256->10) -> d_out =====================
    {
        long long nA = totH;
        k_split_bf16<<<(unsigned)((nA + 255) / 256), 256, 0, stream>>>(big0, ahi, alo, nA);
        k_pack_b<<<dim3(H / 32, NtC), 32, 0, stream>>>(W3, bfh, bfl, H, C);
        k_gemm_bf16x3<<<dim3(gx, NtC), gblock, 0, stream>>>(ahi, alo, bfh, bfl, big1, N, H, C);
    }
    hipMemsetAsync(big2, 0, (size_t)totC * 4, stream);
    k_scatter<<<E, 32, 0, stream>>>(big1, big2, srcI, dstI, dinv, C);
    k_combine<<<(unsigned)((totC + 255) / 256), 256, 0, stream>>>(big2, big1, invdeg, b3,
                                                                  (float*)d_out, totC, C);
}
